// LSTMAEcell_13134009991558
// MI455X (gfx1250) — matrix operand order repaired and checked by reading, not compiled, not run on hardware
//
#include <hip/hip_runtime.h>

// ---------------------------------------------------------------------------
// LSTM autoencoder for MI455X (gfx1250): persistent bf16-WMMA recurrence with
// LDS-resident weights. B=64, T=512, F=256, H=1024, gates G=4096.
//
// Grid: 64 WGs x 128 threads. WG owns h columns [blk*16, blk*16+16) for all 4
// gates; wave w owns batch rows [w*16, w*16+16). Weight slices (4 gates x 16
// cols x K) are staged into LDS once per phase (enc: 162KB, dec: 258KB of the
// 320KB/WGP), so every serial timestep reads B fragments from ds_load_b128
// instead of re-streaming ~10GB of weights from L2 over the sequence.
// ---------------------------------------------------------------------------

#define B_ 64
#define T_ 512
#define F_ 256
#define H_ 1024
#define NWG 64            // workgroups in persistent kernel (H/16)
#define PK_F (F_ + 8)     // padded LDS pitch for K=256  (bank-conflict-free)
#define PK_H (H_ + 8)     // padded LDS pitch for K=1024
#define SMEM_ELEMS (2 * 64 * PK_H)          // decoder phase: two K=1024 slices
#define SMEM_BYTES (SMEM_ELEMS * 2)         // 264,192 B <= 320KB LDS

typedef __attribute__((ext_vector_type(16))) __bf16 v16bf;
typedef __attribute__((ext_vector_type(8)))  float  v8f;

struct __attribute__((aligned(16))) U4 { unsigned int x, y, z, w; };
union Frag { v16bf v; U4 q[2]; };

static __device__ __forceinline__ unsigned short f2bf(float f) {
  unsigned int u = __float_as_uint(f);
  unsigned int r = (u + 0x7FFFu + ((u >> 16) & 1u)) >> 16;  // round-nearest-even
  return (unsigned short)r;
}

static __device__ __forceinline__ float sigmoidf_(float x) {
  return 1.0f / (1.0f + __expf(-x));
}

// ---------------------------------------------------------------------------
// Prep kernels
// ---------------------------------------------------------------------------
__global__ void cvt_bf16_kernel(const float* __restrict__ src,
                                unsigned short* __restrict__ dst, int n) {
  int i = blockIdx.x * 256 + threadIdx.x;
  if (i < n) dst[i] = f2bf(src[i]);
}

// x: [B,T,F] fp32 -> xt: [T,B,F] bf16 (time-major)
__global__ void prep_x_kernel(const float* __restrict__ x,
                              unsigned short* __restrict__ xt) {
  int i = blockIdx.x * 256 + threadIdx.x;
  if (i < T_ * B_ * F_) {
    int f   = i % F_;
    int rem = i / F_;
    int b   = rem % B_;
    int t   = rem / B_;
    xt[i] = f2bf(x[(b * T_ + t) * F_ + f]);
  }
}

__global__ void prep_state_kernel(unsigned short* __restrict__ h0,
                                  float* __restrict__ c,
                                  int* __restrict__ cnt) {
  int i = blockIdx.x * 256 + threadIdx.x;
  if (i < B_ * H_) { h0[i] = 0; c[i] = 0.0f; }
  if (i == 0) *cnt = 0;
}

// ---------------------------------------------------------------------------
// Stage a [4 gates][16 cols][Kdim] weight slice (row-major W[4H][Kdim]) into
// LDS with row pitch Kdim+8 (pad => 4-bank stride across the 16 column lanes).
// ---------------------------------------------------------------------------
static __device__ __forceinline__ void stage_slice(
    const unsigned short* __restrict__ W, int Kdim, int wgcol,
    unsigned short* lds, int tid) {
  const int PK    = Kdim + 8;
  const int total = 64 * Kdim;
  for (int idx = tid * 8; idx < total; idx += 128 * 8) {
    int r = idx / Kdim;          // 0..63 : g*16 + col
    int k = idx - r * Kdim;
    *reinterpret_cast<U4*>(lds + r * PK + k) =
        *reinterpret_cast<const U4*>(
            W + ((r >> 4) * H_ + wgcol + (r & 15)) * Kdim + k);
  }
}

// ---------------------------------------------------------------------------
// gates-tile GEMM: A [B_,K] bf16 (global), B from LDS slice (pitch PK).
// A fragment (16x32): lane L -> row m=L&15; lanes<16 K chunks [0..7],[16..23],
// lanes>=16 [8..15],[24..31]. B fragment (32x16): lane L -> col n=L&15,
// same K chunks as A.
// ---------------------------------------------------------------------------
static __device__ __forceinline__ void gemm_lds(
    const unsigned short* __restrict__ A, int K,
    const unsigned short* __restrict__ Wl, int PK,
    int mbase, int lane, v8f acc[4]) {
  const int row  = lane & 15;   // A row == B col
  const int ksel = lane >> 4;
  const unsigned short* arow = A + (mbase + row) * K;
  for (int kb = 0; kb < K; kb += 32) {
    Frag a;
    const int k0 = kb + ksel * 8;
    a.q[0] = *reinterpret_cast<const U4*>(arow + k0);
    a.q[1] = *reinterpret_cast<const U4*>(arow + k0 + 16);
#pragma unroll
    for (int g = 0; g < 4; ++g) {
      const unsigned short* wrow = Wl + (g * 16 + row) * PK + kb + ksel * 8;
      Frag b;
      b.q[0] = *reinterpret_cast<const U4*>(wrow);
      b.q[1] = *reinterpret_cast<const U4*>(wrow + 16);
      acc[g] = __builtin_amdgcn_wmma_f32_16x16x32_bf16(
          false, a.v, false, b.v, (short)0, acc[g], false, false);
    }
  }
}

// Prefetch next step's A rows (address known one step ahead).
static __device__ __forceinline__ void prefetch_rows(
    const unsigned short* __restrict__ A, int K, int mbase, int lane) {
  const int row  = lane & 15;
  const int ksel = lane >> 4;
  const unsigned short* p = A + (mbase + row) * K + ksel * (K >> 1);
  const int lines = K >> 7;   // 128B cachelines per half-row
#pragma unroll
  for (int i = 0; i < lines; ++i)
    __builtin_prefetch(p + i * 64, 0, 1);
}

// C/D layout: lane L -> n = L&15, VGPR r -> m = r + 8*(L>>4)
static __device__ __forceinline__ void lstm_epilogue(
    v8f acc[4], const float* __restrict__ bias,
    float* __restrict__ c, unsigned short* __restrict__ h_next,
    unsigned short* __restrict__ seq,  // pre-offset by t*B_*H_
    int mbase, int hbase, int lane) {
  int n    = lane & 15;
  int rsel = (lane >> 4) * 8;
  int hidx = hbase + n;
  float bi = bias[0 * H_ + hidx];
  float bf = bias[1 * H_ + hidx];
  float bg = bias[2 * H_ + hidx];
  float bo = bias[3 * H_ + hidx];
#pragma unroll
  for (int r = 0; r < 8; ++r) {
    int off  = (mbase + rsel + r) * H_ + hidx;
    float iv = sigmoidf_(acc[0][r] + bi);
    float fv = sigmoidf_(acc[1][r] + bf);
    float gv = tanhf(acc[2][r] + bg);
    float ov = sigmoidf_(acc[3][r] + bo);
    float cn = fv * c[off] + iv * gv;
    c[off] = cn;
    unsigned short hb = f2bf(ov * tanhf(cn));
    h_next[off] = hb;
    seq[off]    = hb;
  }
}

static __device__ __forceinline__ void grid_sync(int* cnt, int target) {
  __syncthreads();
  if (threadIdx.x == 0) {
    __threadfence();
    atomicAdd(cnt, 1);
    volatile int* v = (volatile int*)cnt;
    while (*v < target) { __builtin_amdgcn_s_sleep(1); }
  }
  __syncthreads();
  __threadfence();
}

// ---------------------------------------------------------------------------
// Persistent encoder+decoder recurrence.
// ---------------------------------------------------------------------------
__global__ __launch_bounds__(128, 1) void lstm_ae_kernel(
    const unsigned short* __restrict__ xt,     // [T,B,F] bf16
    const unsigned short* __restrict__ Wih_e,  // [4H,F] bf16
    const unsigned short* __restrict__ Whh_e,  // [4H,H] bf16
    const float* __restrict__ b_e,
    const unsigned short* __restrict__ Wih_d,  // [4H,H] bf16
    const unsigned short* __restrict__ Whh_d,  // [4H,H] bf16
    const float* __restrict__ b_d,
    unsigned short* __restrict__ h0,           // [B,H] bf16 (ping)
    unsigned short* __restrict__ h1,           // [B,H] bf16 (pong)
    float* __restrict__ c,                     // [B,H] f32
    unsigned short* __restrict__ enc_h,        // [T,B,H] bf16
    unsigned short* __restrict__ dec_h,        // [T,B,H] bf16
    int* __restrict__ cnt) {
  extern __shared__ unsigned short smem[];
  const int tid   = threadIdx.x;
  const int lane  = tid & 31;
  const int mbase = (tid >> 5) * 16;        // batch rows (4 waves cover B=64)
  const int wgcol = blockIdx.x * 16;        // h columns owned by this WG

  const v8f zf = {0.f, 0.f, 0.f, 0.f, 0.f, 0.f, 0.f, 0.f};
  int step = 0;

  // -------- Encoder: stage Wih_e (33KB) + Whh_e (129KB) into LDS --------
  unsigned short* Wih_l = smem;             // pitch PK_F, 64 rows
  unsigned short* Whh_l = smem + 64 * PK_F; // pitch PK_H, 64 rows
  stage_slice(Wih_e, F_, wgcol, Wih_l, tid);
  stage_slice(Whh_e, H_, wgcol, Whh_l, tid);
  __syncthreads();

  for (int t = 0; t < T_; ++t, ++step) {
    const unsigned short* hprev = (step & 1) ? h1 : h0;
    unsigned short*       hnext = (step & 1) ? h0 : h1;
    if (t + 1 < T_) prefetch_rows(xt + (t + 1) * B_ * F_, F_, mbase, lane);
    v8f acc[4];
#pragma unroll
    for (int g = 0; g < 4; ++g) acc[g] = zf;
    gemm_lds(xt + t * B_ * F_, F_, Wih_l, PK_F, mbase, lane, acc);
    gemm_lds(hprev, H_, Whh_l, PK_H, mbase, lane, acc);
    lstm_epilogue(acc, b_e, c, hnext, enc_h + t * B_ * H_, mbase, wgcol, lane);
    grid_sync(cnt, NWG * (step + 1));
  }

  // -------- Decoder: restage Wih_d + Whh_d (258KB), state carries over ----
  unsigned short* Wih_ld = smem;              // pitch PK_H
  unsigned short* Whh_ld = smem + 64 * PK_H;  // pitch PK_H
  stage_slice(Wih_d, H_, wgcol, Wih_ld, tid);
  stage_slice(Whh_d, H_, wgcol, Whh_ld, tid);
  __syncthreads();

  for (int t = 0; t < T_; ++t, ++step) {
    const unsigned short* hprev = (step & 1) ? h1 : h0;
    unsigned short*       hnext = (step & 1) ? h0 : h1;
    if (t + 1 < T_) prefetch_rows(enc_h + (t + 1) * B_ * H_, H_, mbase, lane);
    v8f acc[4];
#pragma unroll
    for (int g = 0; g < 4; ++g) acc[g] = zf;
    gemm_lds(enc_h + t * B_ * H_, H_, Wih_ld, PK_H, mbase, lane, acc);
    gemm_lds(hprev, H_, Whh_ld, PK_H, mbase, lane, acc);
    lstm_epilogue(acc, b_d, c, hnext, dec_h + t * B_ * H_, mbase, wgcol, lane);
    grid_sync(cnt, NWG * (step + 1));
  }
}

// ---------------------------------------------------------------------------
// Output projection: out[b,t,f] = dec_h[t,b,:] @ W_out[f,:] + b_out[f].
// M = T*B = 32768, N = F = 256, K = H = 1024. Fully parallel WMMA GEMM.
// ---------------------------------------------------------------------------
__global__ __launch_bounds__(256, 1) void proj_kernel(
    const unsigned short* __restrict__ dec_h,  // [T*B, H] bf16
    const unsigned short* __restrict__ Wout,   // [F, H] bf16
    const float* __restrict__ b_out,
    float* __restrict__ out) {                 // [B, T, F] f32
  const int lane  = threadIdx.x & 31;
  const int wave  = threadIdx.x >> 5;
  const int mbase = blockIdx.x * 128 + wave * 16;
  const int nbase = blockIdx.y * 64;

  const v8f zf = {0.f, 0.f, 0.f, 0.f, 0.f, 0.f, 0.f, 0.f};
  v8f acc[4];
#pragma unroll
  for (int fi = 0; fi < 4; ++fi) acc[fi] = zf;

  const int row  = lane & 15;
  const int ksel = lane >> 4;
  const unsigned short* arow = dec_h + (mbase + row) * H_;
  for (int kb = 0; kb < H_; kb += 32) {
    Frag a;
    int k0 = kb + ksel * 8;
    a.q[0] = *reinterpret_cast<const U4*>(arow + k0);
    a.q[1] = *reinterpret_cast<const U4*>(arow + k0 + 16);
#pragma unroll
    for (int fi = 0; fi < 4; ++fi) {
      const unsigned short* wrow = Wout + (nbase + fi * 16 + row) * H_ + kb + ksel * 8;
      Frag b;
      b.q[0] = *reinterpret_cast<const U4*>(wrow);
      b.q[1] = *reinterpret_cast<const U4*>(wrow + 16);
      acc[fi] = __builtin_amdgcn_wmma_f32_16x16x32_bf16(
          false, a.v, false, b.v, (short)0, acc[fi], false, false);
    }
  }

  const int n    = lane & 15;
  const int rsel = (lane >> 4) * 8;
#pragma unroll
  for (int fi = 0; fi < 4; ++fi) {
    int col = nbase + fi * 16 + n;
    float bb = b_out[col];
#pragma unroll
    for (int r = 0; r < 8; ++r) {
      int m = mbase + rsel + r;   // row index in [T*B): m = t*B + b
      int t = m >> 6;             // /B_
      int b = m & 63;             // %B_
      out[(b * T_ + t) * F_ + col] = acc[fi][r] + bb;
    }
  }
}

// ---------------------------------------------------------------------------
// Host launcher
// ---------------------------------------------------------------------------
extern "C" void kernel_launch(void* const* d_in, const int* in_sizes, int n_in,
                              void* d_out, int out_size, void* d_ws, size_t ws_size,
                              hipStream_t stream) {
  const float* x     = (const float*)d_in[0];
  const float* Wih_e = (const float*)d_in[1];
  const float* Whh_e = (const float*)d_in[2];
  const float* b_e   = (const float*)d_in[3];
  const float* Wih_d = (const float*)d_in[4];
  const float* Whh_d = (const float*)d_in[5];
  const float* b_d   = (const float*)d_in[6];
  const float* Wout  = (const float*)d_in[7];
  const float* b_out = (const float*)d_in[8];
  (void)n_in; (void)in_sizes; (void)ws_size;

  char* ws = (char*)d_ws;
  size_t off = 0;
  auto carve = [&](size_t bytes) -> char* {
    char* p = ws + off;
    off += (bytes + 255) & ~(size_t)255;
    return p;
  };

  const int nWih_e = 4 * H_ * F_;   // 1,048,576
  const int nWhh   = 4 * H_ * H_;   // 4,194,304
  const int nWout  = F_ * H_;       //   262,144
  const int nX     = T_ * B_ * F_;  // 8,388,608
  const int nState = B_ * H_;       //    65,536
  const int nSeq   = T_ * B_ * H_;  // 33,554,432

  unsigned short* Wih_e_bf = (unsigned short*)carve((size_t)nWih_e * 2);
  unsigned short* Whh_e_bf = (unsigned short*)carve((size_t)nWhh * 2);
  unsigned short* Wih_d_bf = (unsigned short*)carve((size_t)nWhh * 2);
  unsigned short* Whh_d_bf = (unsigned short*)carve((size_t)nWhh * 2);
  unsigned short* Wout_bf  = (unsigned short*)carve((size_t)nWout * 2);
  unsigned short* xt_bf    = (unsigned short*)carve((size_t)nX * 2);
  unsigned short* enc_h    = (unsigned short*)carve((size_t)nSeq * 2);
  unsigned short* dec_h    = (unsigned short*)carve((size_t)nSeq * 2);
  unsigned short* h0       = (unsigned short*)carve((size_t)nState * 2);
  unsigned short* h1       = (unsigned short*)carve((size_t)nState * 2);
  float*          c_st     = (float*)carve((size_t)nState * 4);
  int*            cnt      = (int*)carve(256);

  // Weight / input conversion to bf16 (layouts preserved; W[N][K] row-major
  // is the column-major KxN that the WMMA B fragment wants).
  cvt_bf16_kernel<<<(nWih_e + 255) / 256, 256, 0, stream>>>(Wih_e, Wih_e_bf, nWih_e);
  cvt_bf16_kernel<<<(nWhh + 255) / 256, 256, 0, stream>>>(Whh_e, Whh_e_bf, nWhh);
  cvt_bf16_kernel<<<(nWhh + 255) / 256, 256, 0, stream>>>(Wih_d, Wih_d_bf, nWhh);
  cvt_bf16_kernel<<<(nWhh + 255) / 256, 256, 0, stream>>>(Whh_d, Whh_d_bf, nWhh);
  cvt_bf16_kernel<<<(nWout + 255) / 256, 256, 0, stream>>>(Wout, Wout_bf, nWout);
  prep_x_kernel<<<(nX + 255) / 256, 256, 0, stream>>>(x, xt_bf);
  prep_state_kernel<<<(nState + 255) / 256, 256, 0, stream>>>(h0, c_st, cnt);

  // Persistent recurrence (encoder + decoder), LDS-resident weights,
  // one grid-sync per timestep.
  lstm_ae_kernel<<<NWG, 128, SMEM_BYTES, stream>>>(
      xt_bf, Wih_e_bf, Whh_e_bf, b_e, Wih_d_bf, Whh_d_bf, b_d,
      h0, h1, c_st, enc_h, dec_h, cnt);

  // Output projection GEMM -> d_out [B,T,F] f32.
  dim3 pgrid(256, 4);
  proj_kernel<<<pgrid, 256, 0, stream>>>(dec_h, Wout_bf, b_out, (float*)d_out);
  (void)out_size;
}
